// GATDescriminator_37890201485411
// MI455X (gfx1250) — matrix operand order repaired and checked by reading, not compiled, not run on hardware
//
#include <hip/hip_runtime.h>
#include <hip/hip_bf16.h>
#include <math.h>

typedef __attribute__((ext_vector_type(16))) _Float16 v16h;
typedef __attribute__((ext_vector_type(8)))  _Float16 v8h;
typedef __attribute__((ext_vector_type(8)))  float    v8f;

#define N_NODES 50000
#define N_EDGES 800000
#define E_TOT   (N_EDGES + N_NODES)   // edges + self loops
#define IN_F    128
#define HID     64
#define HEADS   3
#define HC      (HEADS * HID)         // 192
#define NEG_SLOPE   0.2f
#define RRELU_SLOPE ((0.125f + 1.0f / 3.0f) * 0.5f)

__device__ __forceinline__ float leaky(float v, float s) { return v >= 0.f ? v : v * s; }

// monotone float atomic max: int-max for >=0, uint-min for <0 (init to -inf)
__device__ __forceinline__ void atomicMaxF(float* addr, float val) {
    if (val >= 0.f) atomicMax((int*)addr, __float_as_int(val));
    else            atomicMin((unsigned int*)addr, (unsigned int)__float_as_int(val));
}

__device__ __forceinline__ void edge_sd(const int* __restrict__ ei, int e, int& s, int& d) {
    if (e < N_EDGES) { s = ei[e]; d = ei[N_EDGES + e]; }
    else             { s = d = e - N_EDGES; }
}

// ---------------- f32 -> f16 staging ----------------
__global__ void k_cvt_f16(const float* __restrict__ in, _Float16* __restrict__ out, int n) {
    int i = blockIdx.x * blockDim.x + threadIdx.x;
    if (i < n) out[i] = (_Float16)in[i];
}

// ---------------- pack weights f32[K,HC] -> f16 B-fragment order ----------------
// Bp[(((tn*(K/32) + kc)*32 + lane)*16 + j] = (f16) B[(kc*32 + (lane>>4)*16 + j)*HC + tn*16 + (lane&15)]
template <int K>
__global__ void k_pack_B(const float* __restrict__ B, _Float16* __restrict__ Bp) {
    constexpr int KC = K / 32;
    int i = blockIdx.x * blockDim.x + threadIdx.x;
    if (i >= (HC / 16) * KC * 32 * 16) return;
    int j    = i & 15;
    int lane = (i >> 4) & 31;
    int rest = i >> 9;                 // tn*KC + kc
    int tn = rest / KC, kc = rest - tn * KC;
    int k   = kc * 32 + (lane >> 4) * 16 + j;
    int col = tn * 16 + (lane & 15);
    Bp[i] = (_Float16)B[(size_t)k * HC + col];
}

// ---------------- WMMA GEMM: C[M,HC] = A_f16[M,K] * W_f16packed[K,HC] ----------------
// one wave computes a 16x64 slab: A fragment reused across 4 WMMA tiles, K fully unrolled
template <int K>
__global__ void k_wmma_gemm(const _Float16* __restrict__ A, const _Float16* __restrict__ Bp,
                            float* __restrict__ C) {
    constexpr int KC = K / 32;
    int wave = (blockIdx.x * blockDim.x + threadIdx.x) >> 5;
    int lane = threadIdx.x & 31;
    constexpr int rowTiles = N_NODES / 16;        // 3125, exact
    if (wave >= rowTiles * (HC / 64)) return;     // whole-wave exit keeps EXEC all-ones
    int tm = wave / (HC / 64);
    int ng = wave - tm * (HC / 64);               // 0..2, group of 4 N-tiles
    int m0 = tm << 4;
    int hl = lane & 15;                           // row (A) / col (B/C) within tile
    int hi = lane >> 4;                           // lane-half selects K sub-range

    const _Float16* arow = A + (size_t)(m0 + hl) * K;
    v8f acc[4] = {};
#pragma unroll
    for (int kc = 0; kc < KC; kc++) {
        // A 16x32 fragment: lanes 0-15 K=[0..15], lanes 16-31 K=[16..31]
        int ka = kc * 32 + hi * 16;
        v8h alo = *(const v8h*)(arow + ka);
        v8h ahi = *(const v8h*)(arow + ka + 8);
        v16h a;
#pragma unroll
        for (int q = 0; q < 8; q++) { a[q] = alo[q]; a[q + 8] = ahi[q]; }
#pragma unroll
        for (int t = 0; t < 4; t++) {
            int tn = ng * 4 + t;
            const _Float16* bp = Bp + (((size_t)tn * KC + kc) * 32 + lane) * 16;
            v8h blo = *(const v8h*)bp;
            v8h bhi = *(const v8h*)(bp + 8);
            v16h b;
#pragma unroll
            for (int q = 0; q < 8; q++) { b[q] = blo[q]; b[q + 8] = bhi[q]; }
            acc[t] = __builtin_amdgcn_wmma_f32_16x16x32_f16(false, a, false, b, (short)0,
                                                            acc[t], false, false);
        }
    }
    int row0 = m0 + hi * 8;
#pragma unroll
    for (int t = 0; t < 4; t++) {
        int n0 = (ng * 4 + t) * 16;
#pragma unroll
        for (int r = 0; r < 8; r++) C[(size_t)(row0 + r) * HC + n0 + hl] = acc[t][r];
    }
}

// ---------------- per-node attention logits: als/ald[n,h] = <xh[n,h,:], a[h,:]> ----------------
__global__ void k_att_logits(const float* __restrict__ xh, const float* __restrict__ asrc,
                             const float* __restrict__ adst, float* __restrict__ als,
                             float* __restrict__ ald) {
    int i = blockIdx.x * blockDim.x + threadIdx.x;
    if (i >= N_NODES * HEADS) return;
    int node = i / HEADS, h = i - node * HEADS;
    const float* v = xh + (size_t)node * HC + h * HID;
    const float* s = asrc + h * HID;
    const float* d = adst + h * HID;
    float s0 = 0.f, s1 = 0.f;
#pragma unroll 8
    for (int c = 0; c < HID; c++) { float t = v[c]; s0 += t * s[c]; s1 += t * d[c]; }
    als[i] = s0; ald[i] = s1;
}

// ---------------- clear attention state + accumulators ----------------
__global__ void k_init(float* __restrict__ amax, float* __restrict__ denom,
                       float* __restrict__ agg) {
    int i = blockIdx.x * blockDim.x + threadIdx.x;
    if (i < N_NODES * HC) agg[i] = 0.f;
    if (i < N_NODES * HEADS) { amax[i] = -INFINITY; denom[i] = 0.f; }
}

// ---------------- edge pass 1: alpha = leaky(als[src]+ald[dst]); segment max ----------------
__global__ void k_edge_alpha(const int* __restrict__ ei, const float* __restrict__ als,
                             const float* __restrict__ ald, float* __restrict__ alpha,
                             float* __restrict__ amax) {
    int i = blockIdx.x * blockDim.x + threadIdx.x;
    if (i >= E_TOT * HEADS) return;
    int e = i / HEADS, h = i - e * HEADS;
    int s, d; edge_sd(ei, e, s, d);
    float a = leaky(als[s * HEADS + h] + ald[d * HEADS + h], NEG_SLOPE);
    alpha[i] = a;
    atomicMaxF(&amax[d * HEADS + h], a);
}

// ---------------- edge pass 2: ex = exp(alpha - amax[dst]); segment sum ----------------
__global__ void k_edge_exp(const int* __restrict__ ei, float* __restrict__ alpha,
                           const float* __restrict__ amax, float* __restrict__ denom) {
    int i = blockIdx.x * blockDim.x + threadIdx.x;
    if (i >= E_TOT * HEADS) return;
    int e = i / HEADS, h = i - e * HEADS;
    int s, d; edge_sd(ei, e, s, d);
    float ex = __expf(alpha[i] - amax[d * HEADS + h]);
    alpha[i] = ex;
    atomicAdd(&denom[d * HEADS + h], ex);
}

// ---------------- edge pass 3: agg[dst] += xh[src] * ex/denom[dst]  (wave per edge) ----------------
__global__ void k_edge_agg(const int* __restrict__ ei, const float* __restrict__ xh,
                           const float* __restrict__ alpha, const float* __restrict__ denom,
                           float* __restrict__ agg) {
    int wave = (blockIdx.x * blockDim.x + threadIdx.x) >> 5;
    int lane = threadIdx.x & 31;
    if (wave >= E_TOT) return;
    int s, d; edge_sd(ei, wave, s, d);
    const float* xs = xh + (size_t)s * HC;
    float* ad = agg + (size_t)d * HC;
#pragma unroll
    for (int j = 0; j < HC / 32; j++) {
        int hc = j * 32 + lane;
        int h  = hc >> 6;
        float w = alpha[wave * HEADS + h] / (denom[d * HEADS + h] + 1e-16f);
        atomicAdd(&ad[hc], xs[hc] * w);
    }
}

// ---------------- layer-1 finalize: mean over heads + bias + rrelu -> f16 ----------------
__global__ void k_final1(const float* __restrict__ agg, const float* __restrict__ b1,
                         _Float16* __restrict__ h16) {
    int i = blockIdx.x * blockDim.x + threadIdx.x;
    if (i >= N_NODES * HID) return;
    int n = i / HID, c = i - n * HID;
    const float* r = agg + (size_t)n * HC;
    float v = (r[c] + r[HID + c] + r[2 * HID + c]) * (1.f / 3.f) + b1[c];
    h16[i] = (_Float16)leaky(v, RRELU_SLOPE);
}

// ---------------- layer-2 finalize + linear head + sigmoid (wave per node) ----------------
__global__ void k_final_out(const float* __restrict__ agg, const float* __restrict__ b2,
                            const float* __restrict__ Wl, const float* __restrict__ bl,
                            float* __restrict__ out) {
    int wave = (blockIdx.x * blockDim.x + threadIdx.x) >> 5;
    int lane = threadIdx.x & 31;
    if (wave >= N_NODES) return;
    const float* r = agg + (size_t)wave * HC;
    float acc = 0.f;
#pragma unroll
    for (int j = 0; j < HC / 32; j++) {
        int hc = j * 32 + lane;
        acc += leaky(r[hc] + b2[hc], RRELU_SLOPE) * Wl[hc];
    }
#pragma unroll
    for (int off = 16; off; off >>= 1) acc += __shfl_xor(acc, off, 32);
    if (lane == 0) out[wave] = 1.f / (1.f + __expf(-(acc + bl[0])));
}

extern "C" void kernel_launch(void* const* d_in, const int* in_sizes, int n_in,
                              void* d_out, int out_size, void* d_ws, size_t ws_size,
                              hipStream_t stream) {
    (void)in_sizes; (void)n_in; (void)out_size; (void)ws_size;
    const float* x   = (const float*)d_in[0];
    const int*   ei  = (const int*)d_in[1];
    const float* W1  = (const float*)d_in[2];
    const float* a1s = (const float*)d_in[3];
    const float* a1d = (const float*)d_in[4];
    const float* b1  = (const float*)d_in[5];
    const float* W2  = (const float*)d_in[6];
    const float* a2s = (const float*)d_in[7];
    const float* a2d = (const float*)d_in[8];
    const float* b2  = (const float*)d_in[9];
    const float* Wl  = (const float*)d_in[10];
    const float* bl  = (const float*)d_in[11];
    float* out = (float*)d_out;

    char* p = (char*)d_ws;
    auto alloc = [&](size_t bytes) -> char* {
        char* r = p; p += (bytes + 255) & ~(size_t)255; return r;
    };
    float*    xh    = (float*)alloc((size_t)N_NODES * HC * 4);
    float*    agg   = (float*)alloc((size_t)N_NODES * HC * 4);
    float*    als   = (float*)alloc((size_t)N_NODES * HEADS * 4);
    float*    ald   = (float*)alloc((size_t)N_NODES * HEADS * 4);
    float*    amax  = (float*)alloc((size_t)N_NODES * HEADS * 4);
    float*    denom = (float*)alloc((size_t)N_NODES * HEADS * 4);
    float*    alpha = (float*)alloc((size_t)E_TOT * HEADS * 4);
    _Float16* hbuf  = (_Float16*)alloc((size_t)N_NODES * IN_F * 2);
    _Float16* W1p   = (_Float16*)alloc((size_t)IN_F * HC * 2);
    _Float16* W2p   = (_Float16*)alloc((size_t)HID * HC * 2);

    const int B = 256;
    auto cdiv = [](long long a, long long b) { return (int)((a + b - 1) / b); };
    const int slabWaves = (N_NODES / 16) * (HC / 64);   // 3125 * 3 waves (16x64 slabs)

    // ---------------- layer 1 ----------------
    k_cvt_f16<<<cdiv((long long)N_NODES * IN_F, B), B, 0, stream>>>(x, hbuf, N_NODES * IN_F);
    k_pack_B<IN_F><<<cdiv((long long)IN_F * HC, B), B, 0, stream>>>(W1, W1p);
    k_pack_B<HID><<<cdiv((long long)HID * HC, B), B, 0, stream>>>(W2, W2p);
    k_wmma_gemm<IN_F><<<cdiv(slabWaves, 8), 256, 0, stream>>>(hbuf, W1p, xh);
    k_att_logits<<<cdiv((long long)N_NODES * HEADS, B), B, 0, stream>>>(xh, a1s, a1d, als, ald);
    k_init<<<cdiv((long long)N_NODES * HC, B), B, 0, stream>>>(amax, denom, agg);
    k_edge_alpha<<<cdiv((long long)E_TOT * HEADS, B), B, 0, stream>>>(ei, als, ald, alpha, amax);
    k_edge_exp<<<cdiv((long long)E_TOT * HEADS, B), B, 0, stream>>>(ei, alpha, amax, denom);
    k_edge_agg<<<cdiv(E_TOT, 8), 256, 0, stream>>>(ei, xh, alpha, denom, agg);
    k_final1<<<cdiv((long long)N_NODES * HID, B), B, 0, stream>>>(agg, b1, hbuf);

    // ---------------- layer 2 ----------------
    k_wmma_gemm<HID><<<cdiv(slabWaves, 8), 256, 0, stream>>>(hbuf, W2p, xh);
    k_att_logits<<<cdiv((long long)N_NODES * HEADS, B), B, 0, stream>>>(xh, a2s, a2d, als, ald);
    k_init<<<cdiv((long long)N_NODES * HC, B), B, 0, stream>>>(amax, denom, agg);
    k_edge_alpha<<<cdiv((long long)E_TOT * HEADS, B), B, 0, stream>>>(ei, als, ald, alpha, amax);
    k_edge_exp<<<cdiv((long long)E_TOT * HEADS, B), B, 0, stream>>>(ei, alpha, amax, denom);
    k_edge_agg<<<cdiv(E_TOT, 8), 256, 0, stream>>>(ei, xh, alpha, denom, agg);
    k_final_out<<<cdiv(N_NODES, 8), 256, 0, stream>>>(agg, b2, Wl, bl, out);
}
